// SLGAT_32684701122586
// MI455X (gfx1250) — matrix operand order repaired and checked by reading, not compiled, not run on hardware
//
#include <hip/hip_runtime.h>
#include <hip/hip_bf16.h>

typedef _Float16 v16h __attribute__((ext_vector_type(16)));
typedef _Float16 v8h  __attribute__((ext_vector_type(8)));
typedef float    v8f  __attribute__((ext_vector_type(8)));

static constexpr int Nn    = 50000;
static constexpr int Ee    = 800000;
static constexpr int Gg    = 512;
static constexpr int D_IN  = 128;
static constexpr int HID   = 160;
static constexpr int HEADS = 5;
static constexpr int HD    = 32;
static constexpr int OUTC  = 10;

// ---------------------------------------------------------------------------
// Zero a float buffer
__global__ void zero_kernel(float* __restrict__ p, int n) {
    int i = blockIdx.x * blockDim.x + threadIdx.x;
    if (i < n) p[i] = 0.f;
}

// Embedding gather: xe_h[n,d] = emb[x[n], d] (f16)
__global__ void embed_kernel(const int* __restrict__ x, const float* __restrict__ emb,
                             _Float16* __restrict__ xe) {
    int i = blockIdx.x * blockDim.x + threadIdx.x;
    if (i >= Nn * D_IN) return;
    int n = i / D_IN, d = i % D_IN;
    xe[i] = (_Float16)emb[(size_t)x[n] * D_IN + d];
}

// ---------------------------------------------------------------------------
// Weight convert + swizzle into WMMA B-fragment order (CDNA5 ISA 7.12.2, wave32):
// B 32x16 f16 fragment: lane n = lane&15, half = lane>>4; element s holds
// B[kt*32 + 8*half + s + (s>=8 ? 8 : 0)][n].  Stored so each lane's 16 halves are one
// contiguous 32-byte block -> two global_load_b128 in the GEMM hot loop.
__global__ void wswizzle_kernel(const float* __restrict__ B, _Float16* __restrict__ Bsw,
                                int K, int Ncol) {
    int t = blockIdx.x * blockDim.x + threadIdx.x;
    int nT = Ncol >> 4;
    int total = (K >> 5) * nT * 32;
    if (t >= total) return;
    int lane = t & 31;
    int f = t >> 5;
    int kt = f / nT, nt = f % nT;
    int half = lane >> 4;
    int n = nt * 16 + (lane & 15);
    _Float16* o = Bsw + (size_t)t * 16;
#pragma unroll
    for (int v = 0; v < 8; ++v) {
        int k0 = kt * 32 + half * 8 + 2 * v + (v >= 4 ? 8 : 0);
        o[2 * v]     = (_Float16)B[(size_t)k0 * Ncol + n];
        o[2 * v + 1] = (_Float16)B[(size_t)(k0 + 1) * Ncol + n];
    }
}

// ---------------------------------------------------------------------------
// WMMA GEMM: C[M,160] = act(A[M,K] * B[K,160] + bias), f16 in, f32 accumulate.
// One wave32 per 16-row tile.  A-fragments: two aligned b128 loads per K-step
// (layout is K-contiguous per half).  B pre-swizzled to fragment order.
template <int K>
__global__ __launch_bounds__(256)
void gemm_wmma_kernel(const _Float16* __restrict__ A, const _Float16* __restrict__ Bsw,
                      const float* __restrict__ bias, _Float16* __restrict__ Ch,
                      int M, int act) {
    constexpr int KT = K / 32;
    constexpr int NT = HID / 16;          // all GEMMs here have Ncol == 160
    const int lane = threadIdx.x & 31;
    const int wave = blockIdx.x * (blockDim.x >> 5) + (threadIdx.x >> 5);
    const int m0 = wave * 16;
    if (m0 >= M) return;
    const int half = lane >> 4;
    const int mrow = m0 + (lane & 15);

    // Load all A fragments for this 16-row strip once (K-invariant across column tiles)
    v16h afrag[KT];
    const _Float16* arow = A + (size_t)mrow * K + 8 * half;
#pragma unroll
    for (int kt = 0; kt < KT; ++kt) {
        v8h lo = *(const v8h*)(arow + kt * 32);        // K = kt*32 + 8*half + [0..7]
        v8h hi = *(const v8h*)(arow + kt * 32 + 16);   // K = kt*32 + 16 + 8*half + [0..7]
        v16h a;
#pragma unroll
        for (int j = 0; j < 8; ++j) { a[j] = lo[j]; a[8 + j] = hi[j]; }
        afrag[kt] = a;
    }

    const _Float16* bw = Bsw + (size_t)lane * 16;
    for (int nt = 0; nt < NT; ++nt) {
        v8f acc = {};
#pragma unroll
        for (int kt = 0; kt < KT; ++kt) {
            v16h b = *(const v16h*)(bw + (size_t)(kt * NT + nt) * 32 * 16);
            acc = __builtin_amdgcn_wmma_f32_16x16x32_f16(false, afrag[kt], false, b,
                                                         (short)0, acc, false, false);
        }
        const int ncol = nt * 16 + (lane & 15);
        const float bv = bias ? bias[ncol] : 0.f;
#pragma unroll
        for (int r = 0; r < 8; ++r) {
            int row = m0 + half * 8 + r;
            float v = acc[r] + bv;
            if (act == 1) v = v > 0.f ? v : 0.f;       // relu
            Ch[(size_t)row * HID + ncol] = (_Float16)v;
        }
    }
}

// ---------------------------------------------------------------------------
// Structure learner: ew[e] = 0.1*sigmoid(dot(z[src],z[dst])) + 0.1  (wave per edge)
__global__ __launch_bounds__(256)
void edge_sim_kernel(const int* __restrict__ src, const int* __restrict__ dst,
                     const _Float16* __restrict__ z, float* __restrict__ ew) {
    int e = blockIdx.x * (blockDim.x >> 5) + (threadIdx.x >> 5);
    int lane = threadIdx.x & 31;
    if (e >= Ee) return;
    const _Float16* zs = z + (size_t)src[e] * HID;
    const _Float16* zd = z + (size_t)dst[e] * HID;
    float p = 0.f;
#pragma unroll
    for (int k = 0; k < HID / 32; ++k) {
        int d = lane + 32 * k;
        p += (float)zs[d] * (float)zd[d];
    }
    for (int off = 16; off; off >>= 1) p += __shfl_xor(p, off, 32);
    if (lane == 0) {
        float sg = 1.f / (1.f + __expf(-p));
        ew[e] = 0.1f * sg + 0.1f;
    }
}

// Per-node attention scores: s_src[n,h] = sum_hd h[n,h,hd]*a_src[h,hd] (and s_dst)
__global__ void attn_score_kernel(const _Float16* __restrict__ h,
                                  const float* __restrict__ a_src, const float* __restrict__ a_dst,
                                  float* __restrict__ s_src, float* __restrict__ s_dst) {
    int i = blockIdx.x * blockDim.x + threadIdx.x;
    if (i >= Nn * HEADS) return;
    int n = i / HEADS, hh = i % HEADS;
    const _Float16* hp = h + (size_t)n * HID + hh * HD;
    const float* as = a_src + hh * HD;
    const float* ad = a_dst + hh * HD;
    float ss = 0.f, sd = 0.f;
#pragma unroll
    for (int k = 0; k < HD; ++k) {
        float v = (float)hp[k];
        ss += v * as[k];
        sd += v * ad[k];
    }
    s_src[i] = ss;
    s_dst[i] = sd;
}

// Per-layer init: agg = 0, m = -inf, denom = 0
__global__ void init_layer_kernel(float* __restrict__ m, float* __restrict__ denom,
                                  float* __restrict__ agg) {
    int i = blockIdx.x * blockDim.x + threadIdx.x;
    if (i < Nn * HID) agg[i] = 0.f;
    if (i < Nn * HEADS) { m[i] = -__builtin_inff(); denom[i] = 0.f; }
}

__device__ inline void atomicMaxF(float* addr, float val) {
    if (val >= 0.f) atomicMax((int*)addr, __float_as_int(val));
    else            atomicMin((unsigned int*)addr, __float_as_uint(val));
}

// Pass A: e-score = leaky_relu(s_src[src]+s_dst[dst]); segment max into m[dst]
__global__ void pass_a_kernel(const int* __restrict__ src, const int* __restrict__ dst,
                              const float* __restrict__ s_src, const float* __restrict__ s_dst,
                              float* __restrict__ pbuf, float* __restrict__ m) {
    int e = blockIdx.x * blockDim.x + threadIdx.x;
    if (e >= Ee + Nn) return;
    int s, d;
    if (e < Ee) { s = src[e]; d = dst[e]; } else { s = d = e - Ee; }
#pragma unroll
    for (int hh = 0; hh < HEADS; ++hh) {
        float sc = s_src[s * HEADS + hh] + s_dst[d * HEADS + hh];
        sc = sc > 0.f ? sc : 0.2f * sc;
        pbuf[(size_t)e * HEADS + hh] = sc;
        atomicMaxF(&m[d * HEADS + hh], sc);
    }
}

// Pass B: p = exp(e - m[dst]); denom[dst] += p
__global__ void pass_b_kernel(const int* __restrict__ dst, const float* __restrict__ m,
                              float* __restrict__ pbuf, float* __restrict__ denom) {
    int e = blockIdx.x * blockDim.x + threadIdx.x;
    if (e >= Ee + Nn) return;
    int d = (e < Ee) ? dst[e] : (e - Ee);
#pragma unroll
    for (int hh = 0; hh < HEADS; ++hh) {
        float p = __expf(pbuf[(size_t)e * HEADS + hh] - m[d * HEADS + hh]);
        pbuf[(size_t)e * HEADS + hh] = p;
        atomicAdd(&denom[d * HEADS + hh], p);
    }
}

// Pass C: agg[dst] += h[src] * alpha   (wave per edge, lane-strided dims)
__global__ __launch_bounds__(256)
void pass_c_kernel(const int* __restrict__ src, const int* __restrict__ dst,
                   const float* __restrict__ ew, const float* __restrict__ eps,
                   const float* __restrict__ pbuf, const float* __restrict__ denom,
                   const _Float16* __restrict__ hfeat, float* __restrict__ agg) {
    int e = blockIdx.x * (blockDim.x >> 5) + (threadIdx.x >> 5);
    int lane = threadIdx.x & 31;
    if (e >= Ee + Nn) return;
    int s, d; float w;
    if (e < Ee) { s = src[e]; d = dst[e]; w = ew[e]; }
    else        { s = d = e - Ee; w = 1.f + eps[0]; }
    const _Float16* hp = hfeat + (size_t)s * HID;
    float* ap = agg + (size_t)d * HID;
#pragma unroll
    for (int hh = 0; hh < HEADS; ++hh) {
        float alpha = pbuf[(size_t)e * HEADS + hh] /
                      (denom[d * HEADS + hh] + 1e-16f) * w;
        int dim = hh * HD + lane;                       // HD == 32 == wave32
        atomicAdd(&ap[dim], (float)hp[dim] * alpha);
    }
}

// elu(agg + b), optional f16 store of h, atomic per-graph pooling
__global__ void elu_pool_kernel(const float* __restrict__ agg, const float* __restrict__ bvec,
                                const int* __restrict__ batch, _Float16* __restrict__ hh,
                                float* __restrict__ g) {
    int i = blockIdx.x * blockDim.x + threadIdx.x;
    if (i >= Nn * HID) return;
    int n = i / HID, d = i % HID;
    float v = agg[i] + bvec[d];
    v = v > 0.f ? v : __expf(v) - 1.f;
    if (hh) hh[i] = (_Float16)v;
    atomicAdd(&g[(size_t)batch[n] * HID + d], v);
}

// Jumping-knowledge concat [g1|g2] -> f16
__global__ void gcat_kernel(const float* __restrict__ g1, const float* __restrict__ g2,
                            _Float16* __restrict__ gc) {
    int i = blockIdx.x * blockDim.x + threadIdx.x;
    if (i >= Gg * 2 * HID) return;
    int g = i / (2 * HID), c = i % (2 * HID);
    float v = (c < HID) ? g1[g * HID + c] : g2[g * HID + (c - HID)];
    gc[i] = (_Float16)v;
}

// Final tiny layer: out[g,o] = hl[g,:] @ Wl2[:,o] + bl2[o]
__global__ void final_kernel(const _Float16* __restrict__ hl, const float* __restrict__ Wl2,
                             const float* __restrict__ bl2, float* __restrict__ out) {
    int i = blockIdx.x * blockDim.x + threadIdx.x;
    if (i >= Gg * OUTC) return;
    int g = i / OUTC, o = i % OUTC;
    float acc = bl2[o];
#pragma unroll 4
    for (int k = 0; k < HID; ++k)
        acc += (float)hl[(size_t)g * HID + k] * Wl2[k * OUTC + o];
    out[i] = acc;
}

// ---------------------------------------------------------------------------
extern "C" void kernel_launch(void* const* d_in, const int* in_sizes, int n_in,
                              void* d_out, int out_size, void* d_ws, size_t ws_size,
                              hipStream_t stream) {
    const int*   x      = (const int*)d_in[0];
    const int*   ei     = (const int*)d_in[1];
    const int*   batch  = (const int*)d_in[2];
    const float* emb    = (const float*)d_in[3];
    const float* Ws     = (const float*)d_in[4];
    const float* bs     = (const float*)d_in[5];
    const float* eps    = (const float*)d_in[6];
    const float* W1     = (const float*)d_in[7];
    const float* a_src1 = (const float*)d_in[8];
    const float* a_dst1 = (const float*)d_in[9];
    const float* b1     = (const float*)d_in[10];
    const float* W2     = (const float*)d_in[11];
    const float* a_src2 = (const float*)d_in[12];
    const float* a_dst2 = (const float*)d_in[13];
    const float* b2     = (const float*)d_in[14];
    const float* Wj     = (const float*)d_in[15];
    const float* bj     = (const float*)d_in[16];
    const float* Wl1    = (const float*)d_in[17];
    const float* bl1    = (const float*)d_in[18];
    const float* Wl2    = (const float*)d_in[19];
    const float* bl2    = (const float*)d_in[20];
    const int* srcE = ei;
    const int* dstE = ei + Ee;
    float* out = (float*)d_out;

    // Workspace layout (all accumulated buffers re-initialized every call)
    char* ws = (char*)d_ws;
    size_t off = 0;
    auto alloc = [&](size_t bytes) -> char* {
        char* p = ws + off;
        off = (off + bytes + 255) & ~(size_t)255;
        return p;
    };
    _Float16* xe_h   = (_Float16*)alloc((size_t)Nn * D_IN * 2);
    _Float16* z_h    = (_Float16*)alloc((size_t)Nn * HID * 2);
    float*    ew     = (float*)   alloc((size_t)Ee * 4);
    _Float16* hpre_h = (_Float16*)alloc((size_t)Nn * HID * 2);
    float*    ssrc   = (float*)   alloc((size_t)Nn * HEADS * 4);
    float*    sdst   = (float*)   alloc((size_t)Nn * HEADS * 4);
    float*    pbuf   = (float*)   alloc((size_t)(Ee + Nn) * HEADS * 4);
    float*    mmax   = (float*)   alloc((size_t)Nn * HEADS * 4);
    float*    denom  = (float*)   alloc((size_t)Nn * HEADS * 4);
    float*    agg    = (float*)   alloc((size_t)Nn * HID * 4);
    _Float16* h1_h   = (_Float16*)alloc((size_t)Nn * HID * 2);
    float*    g1     = (float*)   alloc((size_t)Gg * HID * 4);
    float*    g2     = (float*)   alloc((size_t)Gg * HID * 4);
    _Float16* gcat_h = (_Float16*)alloc((size_t)Gg * 2 * HID * 2);
    _Float16* hj_h   = (_Float16*)alloc((size_t)Gg * HID * 2);
    _Float16* hl_h   = (_Float16*)alloc((size_t)Gg * HID * 2);
    _Float16* Ws_sw  = (_Float16*)alloc((size_t)D_IN * HID * 2);
    _Float16* W1_sw  = (_Float16*)alloc((size_t)D_IN * HID * 2);
    _Float16* W2_sw  = (_Float16*)alloc((size_t)HID * HID * 2);
    _Float16* Wj_sw  = (_Float16*)alloc((size_t)2 * HID * HID * 2);
    _Float16* Wl1_sw = (_Float16*)alloc((size_t)HID * HID * 2);

    const int T = 256;
    auto blk  = [](int n, int t) { return (n + t - 1) / t; };
    auto wblk = [](int waves)    { return (waves + 7) / 8; };   // 8 wave32 per 256-thr block

    // Weight convert + swizzle into WMMA fragment order (one-time per call, tiny)
    wswizzle_kernel<<<blk((D_IN / 32) * (HID / 16) * 32, T), T, 0, stream>>>(Ws,  Ws_sw,  D_IN, HID);
    wswizzle_kernel<<<blk((D_IN / 32) * (HID / 16) * 32, T), T, 0, stream>>>(W1,  W1_sw,  D_IN, HID);
    wswizzle_kernel<<<blk((HID / 32) * (HID / 16) * 32, T), T, 0, stream>>>(W2,  W2_sw,  HID, HID);
    wswizzle_kernel<<<blk((2 * HID / 32) * (HID / 16) * 32, T), T, 0, stream>>>(Wj, Wj_sw, 2 * HID, HID);
    wswizzle_kernel<<<blk((HID / 32) * (HID / 16) * 32, T), T, 0, stream>>>(Wl1, Wl1_sw, HID, HID);

    // Embedding gather
    embed_kernel<<<blk(Nn * D_IN, T), T, 0, stream>>>(x, emb, xe_h);

    // Structure-learner GEMM: z = relu(xe @ Ws + bs)
    gemm_wmma_kernel<D_IN><<<wblk(Nn / 16), T, 0, stream>>>(xe_h, Ws_sw, bs, z_h, Nn, 1);
    // Layer-1 features: h = xe @ W1
    gemm_wmma_kernel<D_IN><<<wblk(Nn / 16), T, 0, stream>>>(xe_h, W1_sw, nullptr, hpre_h, Nn, 0);

    // Edge weights from similarity
    edge_sim_kernel<<<wblk(Ee), T, 0, stream>>>(srcE, dstE, z_h, ew);

    // Zero pooled outputs
    zero_kernel<<<blk(Gg * HID, T), T, 0, stream>>>(g1, Gg * HID);
    zero_kernel<<<blk(Gg * HID, T), T, 0, stream>>>(g2, Gg * HID);

    // ---- GAT layer 1 ----
    attn_score_kernel<<<blk(Nn * HEADS, T), T, 0, stream>>>(hpre_h, a_src1, a_dst1, ssrc, sdst);
    init_layer_kernel<<<blk(Nn * HID, T), T, 0, stream>>>(mmax, denom, agg);
    pass_a_kernel<<<blk(Ee + Nn, T), T, 0, stream>>>(srcE, dstE, ssrc, sdst, pbuf, mmax);
    pass_b_kernel<<<blk(Ee + Nn, T), T, 0, stream>>>(dstE, mmax, pbuf, denom);
    pass_c_kernel<<<wblk(Ee + Nn), T, 0, stream>>>(srcE, dstE, ew, eps, pbuf, denom, hpre_h, agg);
    elu_pool_kernel<<<blk(Nn * HID, T), T, 0, stream>>>(agg, b1, batch, h1_h, g1);

    // ---- GAT layer 2 ----
    gemm_wmma_kernel<HID><<<wblk(Nn / 16), T, 0, stream>>>(h1_h, W2_sw, nullptr, hpre_h, Nn, 0);
    attn_score_kernel<<<blk(Nn * HEADS, T), T, 0, stream>>>(hpre_h, a_src2, a_dst2, ssrc, sdst);
    init_layer_kernel<<<blk(Nn * HID, T), T, 0, stream>>>(mmax, denom, agg);
    pass_a_kernel<<<blk(Ee + Nn, T), T, 0, stream>>>(srcE, dstE, ssrc, sdst, pbuf, mmax);
    pass_b_kernel<<<blk(Ee + Nn, T), T, 0, stream>>>(dstE, mmax, pbuf, denom);
    pass_c_kernel<<<wblk(Ee + Nn), T, 0, stream>>>(srcE, dstE, ew, eps, pbuf, denom, hpre_h, agg);
    elu_pool_kernel<<<blk(Nn * HID, T), T, 0, stream>>>(agg, b2, batch, nullptr, g2);

    // ---- Head ----
    gcat_kernel<<<blk(Gg * 2 * HID, T), T, 0, stream>>>(g1, g2, gcat_h);
    gemm_wmma_kernel<2 * HID><<<wblk(Gg / 16), T, 0, stream>>>(gcat_h, Wj_sw, bj, hj_h, Gg, 1);
    gemm_wmma_kernel<HID><<<wblk(Gg / 16), T, 0, stream>>>(hj_h, Wl1_sw, bl1, hl_h, Gg, 1);
    final_kernel<<<blk(Gg * OUTC, T), T, 0, stream>>>(hl_h, Wl2, bl2, out);
}
